// BilateralCostVolume_68891275428413
// MI455X (gfx1250) — matrix operand order repaired and checked by reading, not compiled, not run on hardware
//
#include <hip/hip_runtime.h>
#include <hip/hip_bf16.h>

typedef __attribute__((ext_vector_type(16))) _Float16 v16h;
typedef __attribute__((ext_vector_type(8)))  float    v8f;

#define MD_    4
#define RANGE_ 81
#define B_     2
#define C_     48
#define CK_    64          // channels padded to 64 (two 16x16x32 k-steps)
#define H_     80
#define W_     128
#define NPIX_  (B_*H_*W_)  // 20480

// ---------------------------------------------------------------------------
// Pass 1: L2-normalize over the channel dim, convert to f16, transpose to
// channel-last [feat][b][h][w][64] with channels 48..63 zero-padded, so each
// bilinear corner is one contiguous 128B row (8 x b128 loads).
// ---------------------------------------------------------------------------
__global__ void __launch_bounds__(128)
normalize_pack_kernel(const float* __restrict__ f1,
                      const float* __restrict__ f2,
                      _Float16* __restrict__ dst) {
  int idx = blockIdx.x * blockDim.x + threadIdx.x;
  if (idx >= 2 * NPIX_) return;
  int feat = (idx >= NPIX_) ? 1 : 0;
  int p = idx - feat * NPIX_;
  int b = p / (H_ * W_);
  int hw = p % (H_ * W_);
  const float* src = (feat ? f2 : f1) + (size_t)b * C_ * H_ * W_ + hw;
  _Float16* out = dst + (size_t)feat * NPIX_ * CK_ + (size_t)p * CK_;

  float v[C_];
  float ss = 0.f;
#pragma unroll
  for (int c = 0; c < C_; ++c) {          // stride H*W: coalesced across lanes
    v[c] = src[(size_t)c * (H_ * W_)];
    ss += v[c] * v[c];
  }
  float inv = 1.0f / sqrtf(ss + 1e-6f);   // matches reference x/sqrt(ss+1e-6)

  union { float4 f; _Float16 h[8]; } u;
#pragma unroll
  for (int ch = 0; ch < CK_ / 8; ++ch) {
#pragma unroll
    for (int k = 0; k < 8; ++k) {
      int c = ch * 8 + k;
      int cc = (c < C_) ? c : 0;
      u.h[k] = (c < C_) ? (_Float16)(v[cc] * inv) : (_Float16)0.0f;
    }
    ((float4*)out)[ch] = u.f;
  }
}

// ---------------------------------------------------------------------------
// Per-lane bilinear setup: corner row pointer (border clamp), the 4 bilinear
// weights, and the zeros-padding validity mask binarized at 0.999.
// ---------------------------------------------------------------------------
struct Samp {
  const _Float16* row;
  float w0, w1, w2, w3;
  float mask;
};

__device__ __forceinline__ Samp sample_setup(const _Float16* base, int b,
                                             float ix, float iy, int corner) {
  Samp s;
  // ---- features: border padding (clamp before interpolation) ----
  float cx = fminf(fmaxf(ix, 0.f), (float)(W_ - 1));
  float cy = fminf(fmaxf(iy, 0.f), (float)(H_ - 1));
  float fx0 = floorf(cx), fy0 = floorf(cy);
  float wx = cx - fx0, wy = cy - fy0;
  int x0 = (int)fx0, y0 = (int)fy0;
  int x1 = min(x0 + 1, W_ - 1), y1 = min(y0 + 1, H_ - 1);
  int cxI = (corner & 1) ? x1 : x0;
  int cyI = (corner & 2) ? y1 : y0;
  s.row = base + (size_t)((b * H_ + cyI) * W_ + cxI) * CK_;
  s.w0 = (1.f - wx) * (1.f - wy);
  s.w1 = wx * (1.f - wy);
  s.w2 = (1.f - wx) * wy;
  s.w3 = wx * wy;
  // ---- mask: zeros padding on unclamped coords, binarized at 0.999 ----
  float mfx = floorf(ix), mfy = floorf(iy);
  float ax = ix - mfx, ay = iy - mfy;
  float xw = (float)(W_ - 1), yw = (float)(H_ - 1);
  float x0in = (mfx >= 0.f && mfx <= xw) ? 1.f : 0.f;
  float x1in = (mfx + 1.f >= 0.f && mfx + 1.f <= xw) ? 1.f : 0.f;
  float y0in = (mfy >= 0.f && mfy <= yw) ? 1.f : 0.f;
  float y1in = (mfy + 1.f >= 0.f && mfy + 1.f <= yw) ? 1.f : 0.f;
  float m = (1.f - ax) * (1.f - ay) * x0in * y0in
          + ax * (1.f - ay) * x1in * y0in
          + (1.f - ax) * ay * x0in * y1in
          + ax * ay * x1in * y1in;
  s.mask = (m < 0.999f) ? 0.f : 1.f;
  return s;
}

// ---------------------------------------------------------------------------
// Pass 2: one wave per pixel. Each WMMA handles 4 displacements:
//   A row (m = 4q+ca) = f2 forward-warp corner-ca channel row (f16, K=64)
//   B col (n = 4q+cb) = f1 backward-warp corner-cb channel row
//   D[m,n] = sum_c f2fw[ca]*f1bw[cb]   (4x4 diagonal blocks used)
//   cost(r) = maskF*maskB * sum_{ca,cb} wf[ca]*wb[cb]*D[ca,cb]
// ---------------------------------------------------------------------------
__global__ void __launch_bounds__(128)
cost_volume_wmma_kernel(const _Float16* __restrict__ fpk,
                        const float* __restrict__ BM,
                        float* __restrict__ out) {
  const float sW = (float)W_ / (float)(W_ - 1);   // grid_sample scale 128/127
  const float sH = (float)H_ / (float)(H_ - 1);   // 80/79

  int wave = blockIdx.x * (blockDim.x >> 5) + (threadIdx.x >> 5);
  int lane = threadIdx.x & 31;
  int b = wave / (H_ * W_);
  int hw = wave % (H_ * W_);
  int h = hw / W_;
  int w = hw % W_;

  float BMx = BM[(size_t)((b * 2 + 0) * H_ + h) * W_ + w];
  float BMy = BM[(size_t)((b * 2 + 1) * H_ + h) * W_ + w];

  const _Float16* f1n = fpk;                          // backward samples f1
  const _Float16* f2n = fpk + (size_t)NPIX_ * CK_;    // forward samples f2

  int idL = lane & 15;
  int hi = lane >> 4;       // K-half of the row/col this lane carries
  int q = idL >> 2;         // r_local within the group of 4
  int corner = idL & 3;     // bilinear corner this lane loads

  for (int g = 0; g < 21; ++g) {
    int r = 4 * g + q;
    int rc = (r < RANGE_) ? r : RANGE_ - 1;
    float du = (float)(rc % 9 - MD_);
    float dv = (float)(rc / 9 - MD_);

    // t=0.5: forward disp = +(BM+d), backward disp = -(BM+d)
    float ixf = ((float)w + (BMx + du)) * sW - 0.5f;
    float iyf = ((float)h + (BMy + dv)) * sH - 0.5f;
    float ixb = ((float)w - (BMx + du)) * sW - 0.5f;
    float iyb = ((float)h - (BMy + dv)) * sH - 0.5f;

    Samp sf = sample_setup(f2n, b, ixf, iyf, corner);  // A operand rows
    Samp sb = sample_setup(f1n, b, ixb, iyb, corner);  // B operand cols

    const float4* rA = (const float4*)sf.row;          // 8 x 16B per row
    const float4* rB = (const float4*)sb.row;

    union U { v16h v; float4 f[2]; } a0, a1, b0, b1;
    // A layout (16-bit 16x32): lo lanes K{0-7,16-23}, hi lanes K{8-15,24-31}
    int ab = hi;                 // float4 index 0 or 1
    a0.f[0] = rA[ab];     a0.f[1] = rA[ab + 2];
    a1.f[0] = rA[ab + 4]; a1.f[1] = rA[ab + 6];
    // B layout: same as A, lo lanes K{0-7,16-23}, hi lanes K{8-15,24-31}
    int bb = hi;
    b0.f[0] = rB[bb];     b0.f[1] = rB[bb + 2];
    b1.f[0] = rB[bb + 4]; b1.f[1] = rB[bb + 6];

    v8f acc = {};
    acc = __builtin_amdgcn_wmma_f32_16x16x32_f16(false, a0.v, false, b0.v,
                                                 (short)0, acc, false, false);
    acc = __builtin_amdgcn_wmma_f32_16x16x32_f16(false, a1.v, false, b1.v,
                                                 (short)0, acc, false, false);

    // D[4q+ca, n] sits in acc[ca] (q even) / acc[4+ca] (q odd) of the home
    // lane of column n (lo lanes for q<2, hi lanes for q>=2).
    bool odd = (q & 1) != 0;
    float e0 = odd ? acc[4] : acc[0];
    float e1 = odd ? acc[5] : acc[1];
    float e2 = odd ? acc[6] : acc[2];
    float e3 = odd ? acc[7] : acc[3];
    float sA = sf.w0 * e0 + sf.w1 * e1 + sf.w2 * e2 + sf.w3 * e3;
    float wb = (corner == 0) ? sb.w0
             : (corner == 1) ? sb.w1
             : (corner == 2) ? sb.w2 : sb.w3;
    float part = sA * wb;
    // sum the 4 bw-corner columns (lanes differing in low 2 bits)
    part += __shfl_xor(part, 1, 32);
    part += __shfl_xor(part, 2, 32);

    bool reducer = (q < 2) ? (hi == 0) : (hi == 1);
    if (reducer && corner == 0 && r < RANGE_) {
      out[(((size_t)b * RANGE_ + r) * H_ + h) * W_ + w] =
          part * sf.mask * sb.mask;
    }
    // structured flow reconverges here: EXEC is all-ones again before the
    // next iteration's WMMAs.
  }
}

// ---------------------------------------------------------------------------
extern "C" void kernel_launch(void* const* d_in, const int* in_sizes, int n_in,
                              void* d_out, int out_size, void* d_ws,
                              size_t ws_size, hipStream_t stream) {
  (void)in_sizes; (void)n_in; (void)out_size;
  const float* f1 = (const float*)d_in[0];
  const float* f2 = (const float*)d_in[1];
  const float* BM = (const float*)d_in[2];
  float* out = (float*)d_out;

  size_t need = (size_t)2 * NPIX_ * CK_ * sizeof(_Float16);  // 5.25 MB
  if (ws_size < need) return;   // deterministic no-op if scratch too small
  _Float16* fpk = (_Float16*)d_ws;

  int total = 2 * NPIX_;
  normalize_pack_kernel<<<(total + 127) / 128, 128, 0, stream>>>(f1, f2, fpk);
  // one wave per pixel, 4 waves / 128-thread block -> 5120 blocks
  cost_volume_wmma_kernel<<<NPIX_ / 4, 128, 0, stream>>>(fpk, BM, out);
}
